// LSTM_Decoder_2327872275134
// MI455X (gfx1250) — matrix operand order repaired and checked by reading, not compiled, not run on hardware
//
#include <hip/hip_runtime.h>
#include <hip/hip_bf16.h>
#include <math.h>

// ---------------------------------------------------------------------------
// 2-layer LSTM decoder for MI455X (gfx1250, wave32, WMMA).
//   B=128, T=512, IN=1024, H=1024, OUT(fc in)=256
// Design:
//  * gx0 = z @ Wih0^T + b (input is time-constant) computed ONCE.
//  * Persistent kernel: 513 iterations; iteration t does layer0 step t and
//    layer1 step t-1 (pipelined), each as bf16 WMMA GEMMs with f32 accum.
//  * Each wave computes TWO 16x16 row-tiles sharing one B (weight) fragment:
//    halves L2 weight traffic (weights are L2-resident; L2 BW is the binding
//    resource for the serial scan).
//  * h-states ping-pong in two 128x2048 bf16 buffers (L2 resident);
//    c-states live in VGPRs for the whole sequence.
//  * One software grid barrier (L2 atomics) per iteration.
// ---------------------------------------------------------------------------

#define B_   128
#define T_   512
#define H_   1024
#define G4_  4096          // 4*H gate width
#define K1_  2048          // concat(h1, h2) K for layer 1
#define NWG  128           // 4 row blocks x 32 hidden blocks
#define NTHR 256

typedef __attribute__((ext_vector_type(8)))  float        v8f;
typedef __attribute__((ext_vector_type(16))) __bf16       v16bf;
typedef __attribute__((ext_vector_type(4)))  unsigned int u32x4;

union BFrag { u32x4 q[2]; v16bf v; };

static __device__ __forceinline__ unsigned short f2bf(float x) {
  union { float f; unsigned int u; } v; v.f = x;
  unsigned int u = v.u;
  unsigned int r = (u + 0x7fffu + ((u >> 16) & 1u)) >> 16;  // RNE
  return (unsigned short)r;
}
static __device__ __forceinline__ float sigm(float x) {
  return 1.0f / (1.0f + __expf(-x));
}

// ---------------- prep: f32 [R][C] -> bf16 transposed [C + rowOff][4096] ----
__global__ void k_transpose_bf16(const float* __restrict__ src,
                                 unsigned short* __restrict__ dst,
                                 int srcCols, int dstLd, int dstRowOff) {
  __shared__ float tile[32][33];
  const int kx = blockIdx.x * 32, gx = blockIdx.y * 32;
  const int tx = threadIdx.x, ty = threadIdx.y;
#pragma unroll
  for (int i = 0; i < 4; ++i)
    tile[ty + i * 8][tx] = src[(long)(gx + ty + i * 8) * srcCols + kx + tx];
  __syncthreads();
#pragma unroll
  for (int i = 0; i < 4; ++i)
    dst[(long)(kx + ty + i * 8 + dstRowOff) * dstLd + gx + tx] =
        f2bf(tile[tx][ty + i * 8]);
}

__global__ void k_bias(const float* __restrict__ a, const float* __restrict__ b,
                       float* __restrict__ c) {
  int i = blockIdx.x * 256 + threadIdx.x;
  if (i < G4_) c[i] = a[i] + b[i];
}

// ---------------- prep: z = x @ fc_w^T + fc_b  ([128,1024], K=256) ----------
__global__ void k_fc(const float* __restrict__ x, const float* __restrict__ fcw,
                     const float* __restrict__ fcb, float* __restrict__ z) {
  __shared__ float xs[256];
  const int b = blockIdx.y;
  const int i = blockIdx.x * 256 + threadIdx.x;
  xs[threadIdx.x] = x[b * 256 + threadIdx.x];
  __syncthreads();
  float acc = fcb[i];
#pragma unroll 8
  for (int o = 0; o < 256; ++o) acc += xs[o] * fcw[i * 256 + o];
  z[b * 1024 + i] = acc;
}

// ---------------- prep: gx0 = z @ Wih0^T + bih0 + bhh0  ([128,4096]) --------
__global__ void k_gx0(const float* __restrict__ z, const float* __restrict__ Wih0,
                      const float* __restrict__ bih0, const float* __restrict__ bhh0,
                      float* __restrict__ gx0) {
  __shared__ float Zs[32][33], Ws[32][33];
  const int gb = blockIdx.x * 32, rb = blockIdx.y * 32;
  const int tx = threadIdx.x, ty = threadIdx.y;
  float acc[4] = {0.f, 0.f, 0.f, 0.f};
  for (int k0 = 0; k0 < H_; k0 += 32) {
#pragma unroll
    for (int i = 0; i < 4; ++i) {
      Zs[ty + i * 8][tx] = z[(rb + ty + i * 8) * H_ + k0 + tx];
      Ws[ty + i * 8][tx] = Wih0[(long)(gb + ty + i * 8) * H_ + k0 + tx];
    }
    __syncthreads();
#pragma unroll 8
    for (int kk = 0; kk < 32; ++kk) {
      float w = Ws[tx][kk];
#pragma unroll
      for (int i = 0; i < 4; ++i) acc[i] += Zs[ty + i * 8][kk] * w;
    }
    __syncthreads();
  }
  const int g = gb + tx;
  const float bias = bih0[g] + bhh0[g];
#pragma unroll
  for (int i = 0; i < 4; ++i)
    gx0[(long)(rb + ty + i * 8) * G4_ + g] = acc[i] + bias;
}

// ---------------- persistent LSTM scan --------------------------------------
__global__ __launch_bounds__(NTHR, 1)
void k_lstm(const unsigned short* __restrict__ Wt0,   // [1024][4096] bf16 = Whh0^T
            const unsigned short* __restrict__ W1t,   // [2048][4096] bf16 = [Wih1;Whh1]^T
            const float* __restrict__ gx0,            // [128][4096]
            const float* __restrict__ b1,             // [4096] bih1+bhh1
            unsigned short* __restrict__ hbA,         // [128][2048] bf16 ping
            unsigned short* __restrict__ hbB,         // [128][2048] bf16 pong
            float* __restrict__ out,                  // [128][512][1024]
            unsigned int* __restrict__ bar) {
  __shared__ float lds[2][4][32][33];                 // [layer][gate][row][col(+pad)]

  const int wgid = blockIdx.x;
  const int rb   = wgid >> 5;          // 0..3   : 32-row block
  const int hb   = wgid & 31;          // 0..31  : 32-hidden-col block
  const int tid  = threadIdx.x;
  const int wave = tid >> 5;
  const int lane = tid & 31;
  const int gsel = wave >> 1;          // gate 0..3 (i,f,g,o)
  const int ch   = wave & 1;           // column half of the 32-col block
  const int half = lane >> 4;
  const int l15  = lane & 15;
  const int rowBase = rb * 32;
  const int mRow0 = rowBase + l15;            // A row, tile 0 (rows 0..15)
  const int mRow1 = rowBase + 16 + l15;       // A row, tile 1 (rows 16..31)
  const int n0    = gsel * H_ + hb * 32 + ch * 16;  // gate column base

  const int er = tid >> 3;             // elementwise row 0..31
  const int ec = (tid & 7) * 4;        // elementwise col base (4 cols/thread)

  float c0s[4] = {0.f, 0.f, 0.f, 0.f};  // layer0 cell state (VGPRs, whole scan)
  float c2s[4] = {0.f, 0.f, 0.f, 0.f};  // layer1 cell state

  for (int t = 0; t <= T_; ++t) {
    const unsigned short* bufIn  = (t & 1) ? hbB : hbA;
    unsigned short*       bufOut = (t & 1) ? hbA : hbB;

    // ---- layer 0 GEMM: G0 = h0 @ Whh0^T (two 16x16 row tiles share B) ----
    if (t < T_) {
      v8f acc0 = {0.f, 0.f, 0.f, 0.f, 0.f, 0.f, 0.f, 0.f};
      v8f acc1 = {0.f, 0.f, 0.f, 0.f, 0.f, 0.f, 0.f, 0.f};
#pragma unroll 2
      for (int k0 = 0; k0 < H_; k0 += 32) {
        BFrag a0, a1, b;
        const unsigned short* ap0 = bufIn + (long)mRow0 * K1_ + k0 + half * 8;
        const unsigned short* ap1 = bufIn + (long)mRow1 * K1_ + k0 + half * 8;
        a0.q[0] = *(const u32x4*)(ap0);       // K chunk [k0+half*8, +8)
        a0.q[1] = *(const u32x4*)(ap0 + 16);  // K chunk [k0+16+half*8, +8)
        a1.q[0] = *(const u32x4*)(ap1);
        a1.q[1] = *(const u32x4*)(ap1 + 16);
        const unsigned short* bp = Wt0 + (long)(k0 + half * 8) * G4_ + n0 + l15;
#pragma unroll
        for (int j = 0; j < 4; ++j) {
          b.q[0][j] = (unsigned)bp[(long)(2 * j) * G4_] | ((unsigned)bp[(long)(2 * j + 1) * G4_] << 16);        // K [k0+half*8, +8)
          b.q[1][j] = (unsigned)bp[(long)(16 + 2 * j) * G4_] | ((unsigned)bp[(long)(17 + 2 * j) * G4_] << 16);  // K [k0+16+half*8, +8)
        }
        acc0 = __builtin_amdgcn_wmma_f32_16x16x32_bf16(
            false, a0.v, false, b.v, (short)0, acc0, false, false);
        acc1 = __builtin_amdgcn_wmma_f32_16x16x32_bf16(
            false, a1.v, false, b.v, (short)0, acc1, false, false);
      }
#pragma unroll
      for (int r = 0; r < 8; ++r) {
        lds[0][gsel][half * 8 + r][ch * 16 + l15]      = acc0[r];
        lds[0][gsel][16 + half * 8 + r][ch * 16 + l15] = acc1[r];
      }
    }

    // ---- layer 1 GEMM: G1 = [h1,h2] @ [Wih1;Whh1]^T (K=2048), step t-1 ----
    if (t >= 1) {
      v8f acc0 = {0.f, 0.f, 0.f, 0.f, 0.f, 0.f, 0.f, 0.f};
      v8f acc1 = {0.f, 0.f, 0.f, 0.f, 0.f, 0.f, 0.f, 0.f};
#pragma unroll 2
      for (int k0 = 0; k0 < K1_; k0 += 32) {
        BFrag a0, a1, b;
        const unsigned short* ap0 = bufIn + (long)mRow0 * K1_ + k0 + half * 8;
        const unsigned short* ap1 = bufIn + (long)mRow1 * K1_ + k0 + half * 8;
        a0.q[0] = *(const u32x4*)(ap0);
        a0.q[1] = *(const u32x4*)(ap0 + 16);
        a1.q[0] = *(const u32x4*)(ap1);
        a1.q[1] = *(const u32x4*)(ap1 + 16);
        const unsigned short* bp = W1t + (long)(k0 + half * 8) * G4_ + n0 + l15;
#pragma unroll
        for (int j = 0; j < 4; ++j) {
          b.q[0][j] = (unsigned)bp[(long)(2 * j) * G4_] | ((unsigned)bp[(long)(2 * j + 1) * G4_] << 16);
          b.q[1][j] = (unsigned)bp[(long)(16 + 2 * j) * G4_] | ((unsigned)bp[(long)(17 + 2 * j) * G4_] << 16);
        }
        acc0 = __builtin_amdgcn_wmma_f32_16x16x32_bf16(
            false, a0.v, false, b.v, (short)0, acc0, false, false);
        acc1 = __builtin_amdgcn_wmma_f32_16x16x32_bf16(
            false, a1.v, false, b.v, (short)0, acc1, false, false);
      }
#pragma unroll
      for (int r = 0; r < 8; ++r) {
        lds[1][gsel][half * 8 + r][ch * 16 + l15]      = acc0[r];
        lds[1][gsel][16 + half * 8 + r][ch * 16 + l15] = acc1[r];
      }
    }

    __syncthreads();

    // ---- layer 0 cell update: h1[t] -> bufOut[:, 0:1024] -------------------
    if (t < T_) {
#pragma unroll
      for (int j = 0; j < 4; ++j) {
        const int c = ec + j;
        const int gcol = hb * 32 + c;
        const long gxb = (long)(rowBase + er) * G4_ + gcol;
        float iv = lds[0][0][er][c] + gx0[gxb];
        float fv = lds[0][1][er][c] + gx0[gxb + 1024];
        float gv = lds[0][2][er][c] + gx0[gxb + 2048];
        float ov = lds[0][3][er][c] + gx0[gxb + 3072];
        float cc = sigm(fv) * c0s[j] + sigm(iv) * tanhf(gv);
        c0s[j] = cc;
        float h = sigm(ov) * tanhf(cc);
        bufOut[(long)(rowBase + er) * K1_ + gcol] = f2bf(h);
      }
    }

    // ---- layer 1 cell update (step t-1): h2 -> bufOut[:,1024:2048] + d_out -
    if (t >= 1) {
#pragma unroll
      for (int j = 0; j < 4; ++j) {
        const int c = ec + j;
        const int gcol = hb * 32 + c;
        float iv = lds[1][0][er][c] + b1[gcol];
        float fv = lds[1][1][er][c] + b1[1024 + gcol];
        float gv = lds[1][2][er][c] + b1[2048 + gcol];
        float ov = lds[1][3][er][c] + b1[3072 + gcol];
        float cc = sigm(fv) * c2s[j] + sigm(iv) * tanhf(gv);
        c2s[j] = cc;
        float h = sigm(ov) * tanhf(cc);
        bufOut[(long)(rowBase + er) * K1_ + H_ + gcol] = f2bf(h);
        out[((long)(rowBase + er) * T_ + (t - 1)) * H_ + gcol] = h;
      }
    }

    // ---- device-wide barrier (sense via generation counter) ----------------
    __threadfence();
    __syncthreads();
    if (tid == 0) {
      unsigned my = __hip_atomic_load(&bar[1], __ATOMIC_RELAXED,
                                      __HIP_MEMORY_SCOPE_AGENT);
      unsigned prev = __hip_atomic_fetch_add(&bar[0], 1u, __ATOMIC_ACQ_REL,
                                             __HIP_MEMORY_SCOPE_AGENT);
      if (prev == NWG - 1) {
        __hip_atomic_store(&bar[0], 0u, __ATOMIC_RELAXED,
                           __HIP_MEMORY_SCOPE_AGENT);
        __hip_atomic_fetch_add(&bar[1], 1u, __ATOMIC_RELEASE,
                               __HIP_MEMORY_SCOPE_AGENT);
      } else {
        while (__hip_atomic_load(&bar[1], __ATOMIC_ACQUIRE,
                                 __HIP_MEMORY_SCOPE_AGENT) == my)
          __builtin_amdgcn_s_sleep(8);
      }
    }
    __syncthreads();
    __threadfence();  // invalidate L0 so next iter sees other WGs' h writes
  }
}

// ---------------------------------------------------------------------------
extern "C" void kernel_launch(void* const* d_in, const int* in_sizes, int n_in,
                              void* d_out, int out_size, void* d_ws, size_t ws_size,
                              hipStream_t stream) {
  (void)in_sizes; (void)n_in; (void)out_size; (void)ws_size;
  const float* x    = (const float*)d_in[0];
  const float* fcw  = (const float*)d_in[1];
  const float* fcb  = (const float*)d_in[2];
  const float* Wih0 = (const float*)d_in[3];
  const float* Whh0 = (const float*)d_in[4];
  const float* bih0 = (const float*)d_in[5];
  const float* bhh0 = (const float*)d_in[6];
  const float* Wih1 = (const float*)d_in[7];
  const float* Whh1 = (const float*)d_in[8];
  const float* bih1 = (const float*)d_in[9];
  const float* bhh1 = (const float*)d_in[10];
  float* out = (float*)d_out;

  char* ws = (char*)d_ws;
  size_t off = 0;
  auto alloc = [&](size_t bytes) -> void* {
    void* p = ws + off;
    off += (bytes + 255) & ~size_t(255);
    return p;
  };
  unsigned int*   bar = (unsigned int*)  alloc(256);
  unsigned short* Wt0 = (unsigned short*)alloc((size_t)H_  * G4_ * 2);  // 8MB
  unsigned short* W1t = (unsigned short*)alloc((size_t)K1_ * G4_ * 2);  // 16MB
  float*          gx0 = (float*)         alloc((size_t)B_ * G4_ * 4);   // 2MB
  float*          b1  = (float*)         alloc((size_t)G4_ * 4);
  float*          z   = (float*)         alloc((size_t)B_ * H_ * 4);
  unsigned short* hbA = (unsigned short*)alloc((size_t)B_ * K1_ * 2);
  unsigned short* hbB = (unsigned short*)alloc((size_t)B_ * K1_ * 2);

  hipMemsetAsync(bar, 0, 256, stream);
  hipMemsetAsync(hbA, 0, (size_t)B_ * K1_ * 2, stream);
  hipMemsetAsync(hbB, 0, (size_t)B_ * K1_ * 2, stream);

  dim3 tb(32, 8);
  // Wt0[k][g] = Whh0[g][k]; W1t[k][g] = Wih1[g][k]; W1t[1024+k][g] = Whh1[g][k]
  k_transpose_bf16<<<dim3(H_ / 32, G4_ / 32), tb, 0, stream>>>(Whh0, Wt0, H_, G4_, 0);
  k_transpose_bf16<<<dim3(H_ / 32, G4_ / 32), tb, 0, stream>>>(Wih1, W1t, H_, G4_, 0);
  k_transpose_bf16<<<dim3(H_ / 32, G4_ / 32), tb, 0, stream>>>(Whh1, W1t, H_, G4_, H_);
  k_bias<<<G4_ / 256, 256, 0, stream>>>(bih1, bhh1, b1);
  k_fc<<<dim3(H_ / 256, B_), 256, 0, stream>>>(x, fcw, fcb, z);
  k_gx0<<<dim3(G4_ / 32, B_ / 32), tb, 0, stream>>>(z, Wih0, bih0, bhh0, gx0);
  k_lstm<<<NWG, NTHR, 0, stream>>>(Wt0, W1t, gx0, b1, hbA, hbB, out, bar);
}
